// Mamba2Block_67774583931209
// MI455X (gfx1250) — matrix operand order repaired and checked by reading, not compiled, not run on hardware
//
#include <hip/hip_runtime.h>
#include <hip/hip_bf16.h>
#include <math.h>

// ---------------------------------------------------------------------------
// Mamba2 block forward for MI455X (gfx1250, wave32, WMMA bf16 path)
// ---------------------------------------------------------------------------

typedef __bf16 bf16_t;
typedef __attribute__((ext_vector_type(16))) __bf16 v16bf;
typedef __attribute__((ext_vector_type(8)))  __bf16 v8bf;
typedef __attribute__((ext_vector_type(8)))  float  v8f;

#define DIM      2048
#define D_STATE  128
#define D_CONV   4
#define N_HEADS  48
#define D_HEAD   64
#define D_INNER  3072          // N_HEADS * D_HEAD
#define D_PROJ   6448          // 2*D_INNER + 2*D_STATE + N_HEADS
#define D_PROJ_PAD 6464        // padded to multiple of 32 for the GEMM N-tile
#define SEQ      1024
#define EPS      1e-5f

// proj column offsets
#define OFF_Z    0
#define OFF_XC   D_INNER                 // 3072
#define OFF_B    (2*D_INNER)             // 6144
#define OFF_C    (2*D_INNER + D_STATE)   // 6272
#define OFF_DT   (2*D_INNER + 2*D_STATE) // 6400

// ---------------------------------------------------------------------------
// 1) RMSNorm over DIM, emit bf16 activations for the WMMA GEMM
// ---------------------------------------------------------------------------
__global__ void rmsnorm_x_kernel(const float* __restrict__ x,
                                 const float* __restrict__ w,
                                 bf16_t* __restrict__ xn_bf) {
    const int s = blockIdx.x;
    const int tid = threadIdx.x;
    const float* row = x + (size_t)s * DIM;

    float ss = 0.f;
#pragma unroll
    for (int j = 0; j < DIM / 256; ++j) {
        float v = row[tid + j * 256];
        ss += v * v;
    }
    __shared__ float red[256];
    red[tid] = ss;
    __syncthreads();
    for (int off = 128; off > 0; off >>= 1) {
        if (tid < off) red[tid] += red[tid + off];
        __syncthreads();
    }
    const float inv = rsqrtf(red[0] / (float)DIM + EPS);
#pragma unroll
    for (int j = 0; j < DIM / 256; ++j) {
        int c = tid + j * 256;
        xn_bf[(size_t)s * DIM + c] = (bf16_t)(row[c] * inv * w[c]);
    }
}

// ---------------------------------------------------------------------------
// 2) fp32 -> bf16 weight conversion with row padding (pad rows -> 0)
// ---------------------------------------------------------------------------
__global__ void cvt_bf16_kernel(const float* __restrict__ src,
                                bf16_t* __restrict__ dst,
                                int rows, int cols, int pad_rows) {
    int i = blockIdx.x * 256 + threadIdx.x;
    if (i >= pad_rows * cols) return;
    int r = i / cols;
    dst[i] = (r < rows) ? (bf16_t)src[i] : (bf16_t)0.f;
}

// ---------------------------------------------------------------------------
// 3) NT GEMM on the WMMA path:  C[M,N] = A[M,K] * B[Npad,K]^T  (+ residual)
//    A,B bf16 row-major; C fp32.
//    Wave tile: 64(M) x 32(N) = 8 accumulators; 4 A-fragments x 2 B-fragments
//    per 32-wide K step (8 WMMA / 12 b128 loads). Register double buffer:
//    next K-tile fragments are loaded before the current tile's WMMAs.
//    Block = 128 threads = 4 waves stacked along M (256 rows / block).
// ---------------------------------------------------------------------------
__global__ __launch_bounds__(128)
void gemm_bf16_nt_kernel(const bf16_t* __restrict__ A,
                         const bf16_t* __restrict__ B,
                         const float*  __restrict__ residual,  // may be null
                         float* __restrict__ C,
                         int M, int N, int K) {
    const int lane = threadIdx.x & 31;
    const int wave = threadIdx.x >> 5;
    const int n0   = blockIdx.x * 32;
    const int m0   = blockIdx.y * 256 + wave * 64;

    const int l16  = lane & 15;
    const int half = lane >> 4;          // 0 or 1

    union AFrag { v16bf v; v8bf h[2]; };

    // per-lane base pointers
    const bf16_t* arow[4];
#pragma unroll
    for (int i = 0; i < 4; ++i)
        arow[i] = A + (size_t)(m0 + i * 16 + l16) * K + half * 16;
    const bf16_t* brow[2];
#pragma unroll
    for (int j = 0; j < 2; ++j)
        brow[j] = B + (size_t)(n0 + j * 16 + l16) * K + half * 16;

    v8f acc[4][2];
#pragma unroll
    for (int i = 0; i < 4; ++i)
#pragma unroll
        for (int j = 0; j < 2; ++j)
            acc[i][j] = (v8f){};

    // prologue: load K-tile 0
    AFrag a[4];
    v16bf b[2];
#pragma unroll
    for (int i = 0; i < 4; ++i) {
        a[i].h[0] = *(const v8bf*)(arow[i]);
        a[i].h[1] = *(const v8bf*)(arow[i] + 8);
    }
#pragma unroll
    for (int j = 0; j < 2; ++j)
        b[j] = *(const v16bf*)(brow[j]);

    for (int k0 = 0; k0 < K; k0 += 32) {
        // issue next tile's loads before using this tile (SW pipeline)
        const int kn = (k0 + 32 < K) ? (k0 + 32) : 0;   // last iter: harmless reload
        AFrag an[4];
        v16bf bn[2];
#pragma unroll
        for (int i = 0; i < 4; ++i) {
            an[i].h[0] = *(const v8bf*)(arow[i] + kn);
            an[i].h[1] = *(const v8bf*)(arow[i] + kn + 8);
        }
#pragma unroll
        for (int j = 0; j < 2; ++j) {
            bn[j] = *(const v16bf*)(brow[j] + kn);
            __builtin_prefetch(brow[j] + kn + 32, 0, 1);
        }

        // 8 WMMAs on the current fragments
#pragma unroll
        for (int i = 0; i < 4; ++i)
#pragma unroll
            for (int j = 0; j < 2; ++j)
                acc[i][j] = __builtin_amdgcn_wmma_f32_16x16x32_bf16(
                    false, a[i].v, false, b[j], (short)0, acc[i][j], false, false);

        // rotate buffers
#pragma unroll
        for (int i = 0; i < 4; ++i) a[i] = an[i];
        b[0] = bn[0];
        b[1] = bn[1];
    }

    // epilogue: C/D layout -> lane holds column n = l16, rows v + 8*half
    const int mb = half * 8;
#pragma unroll
    for (int j = 0; j < 2; ++j) {
        const int n = n0 + j * 16 + l16;
        if (n >= N) continue;
#pragma unroll
        for (int i = 0; i < 4; ++i) {
#pragma unroll
            for (int v = 0; v < 8; ++v) {
                const int m = m0 + i * 16 + mb + v;
                float val = acc[i][j][v];
                if (residual) val += residual[(size_t)m * N + n];
                C[(size_t)m * N + n] = val;
            }
        }
    }
}

// ---------------------------------------------------------------------------
// 4) depthwise causal conv (width 4) + SiLU  -> x_ssm [SEQ, D_INNER]
// ---------------------------------------------------------------------------
__global__ void conv_silu_kernel(const float* __restrict__ proj,
                                 const float* __restrict__ conv_w,
                                 const float* __restrict__ conv_b,
                                 const float* __restrict__ conv_state,
                                 float* __restrict__ x_ssm) {
    const int c = blockIdx.x * 256 + threadIdx.x;   // channel 0..3071
    const int s = blockIdx.y;                       // time step
    if (c >= D_INNER) return;

    float v = conv_b[c];
#pragma unroll
    for (int k = 0; k < D_CONV; ++k) {
        const int t = s - (D_CONV - 1) + k;
        float xk = (t >= 0) ? proj[(size_t)t * D_PROJ + OFF_XC + c]
                            : conv_state[(size_t)c * (D_CONV - 1) + (t + D_CONV - 1)];
        v += conv_w[c * D_CONV + k] * xk;
    }
    v = v / (1.f + expf(-v));                       // SiLU
    x_ssm[(size_t)s * D_INNER + c] = v;
}

// ---------------------------------------------------------------------------
// 5) sequential SSM scan. Parallelism: 48 heads x 2 p-halves = 96 WGPs.
//    thread t: p_local = t&31 (p = pbase+p_local), n-chunk = (t>>5)*16.
//    State kept in registers (16 floats/thread); B/C staged in LDS per step.
// ---------------------------------------------------------------------------
__global__ __launch_bounds__(256)
void ssm_scan_kernel(const float* __restrict__ proj,
                     const float* __restrict__ x_ssm,
                     const float* __restrict__ dt_bias,
                     const float* __restrict__ A_log,
                     const float* __restrict__ D_param,
                     const float* __restrict__ ssm_state0,
                     float* __restrict__ y_out) {
    const int head  = blockIdx.x >> 1;
    const int pbase = (blockIdx.x & 1) * 32;
    const int tid   = threadIdx.x;
    const int pl    = tid & 31;       // local p
    const int p     = pbase + pl;     // global p in head
    const int ng    = tid >> 5;       // 0..7
    const int n0    = ng * 16;

    __shared__ float sB[D_STATE];
    __shared__ float sC[D_STATE];
    __shared__ float sx[32];
    __shared__ float s_dA;
    __shared__ float spart[8 * 32];

    const float A  = -expf(A_log[head]);
    const float Dp = D_param[head];
    const float db = dt_bias[head];

    float h[16];
#pragma unroll
    for (int j = 0; j < 16; ++j)
        h[j] = ssm_state0[(size_t)head * D_STATE * D_HEAD + (size_t)(n0 + j) * D_HEAD + p];

    for (int s = 0; s < SEQ; ++s) {
        const float* prow = proj + (size_t)s * D_PROJ;
        if (tid < D_STATE)            sB[tid] = prow[OFF_B + tid];
        else if (tid < 2 * D_STATE)   sC[tid - D_STATE] = prow[OFF_C + (tid - D_STATE)];
        if (tid < 32)
            sx[tid] = x_ssm[(size_t)s * D_INNER + head * D_HEAD + pbase + tid];
        if (tid == 0) {
            float dtr = prow[OFF_DT + head] + db;
            float dt  = (dtr > 20.f) ? dtr : log1pf(expf(dtr));   // softplus
            s_dA = expf(A * dt);
        }
        __syncthreads();

        const float xp = sx[pl];
        const float dA = s_dA;
        float part = 0.f;
#pragma unroll
        for (int j = 0; j < 16; ++j) {
            const int n = n0 + j;
            h[j] = dA * h[j] + sB[n] * xp;
            part += sC[n] * h[j];
        }
        spart[ng * 32 + pl] = part;
        __syncthreads();

        if (ng == 0) {
            float y = Dp * xp;
#pragma unroll
            for (int g = 0; g < 8; ++g) y += spart[g * 32 + pl];
            y_out[(size_t)s * D_INNER + head * D_HEAD + p] = y;
        }
    }
}

// ---------------------------------------------------------------------------
// 6) inner RMSNorm over D_INNER + SiLU gate, emit bf16 for out_proj GEMM
// ---------------------------------------------------------------------------
__global__ void gated_norm_kernel(const float* __restrict__ y,
                                  const float* __restrict__ proj,
                                  const float* __restrict__ w,
                                  bf16_t* __restrict__ yg_bf) {
    const int s = blockIdx.x;
    const int tid = threadIdx.x;
    const float* row = y + (size_t)s * D_INNER;

    float ss = 0.f;
#pragma unroll
    for (int j = 0; j < D_INNER / 256; ++j) {
        float v = row[tid + j * 256];
        ss += v * v;
    }
    __shared__ float red[256];
    red[tid] = ss;
    __syncthreads();
    for (int off = 128; off > 0; off >>= 1) {
        if (tid < off) red[tid] += red[tid + off];
        __syncthreads();
    }
    const float inv = rsqrtf(red[0] / (float)D_INNER + EPS);
#pragma unroll
    for (int j = 0; j < D_INNER / 256; ++j) {
        int c = tid + j * 256;
        float z = proj[(size_t)s * D_PROJ + OFF_Z + c];
        float gate = z / (1.f + expf(-z));           // SiLU(z)
        yg_bf[(size_t)s * D_INNER + c] = (bf16_t)(row[c] * inv * w[c] * gate);
    }
}

// ---------------------------------------------------------------------------
// launcher
// ---------------------------------------------------------------------------
extern "C" void kernel_launch(void* const* d_in, const int* in_sizes, int n_in,
                              void* d_out, int out_size, void* d_ws, size_t ws_size,
                              hipStream_t stream) {
    const float* x          = (const float*)d_in[0];   // [1,1024,2048]
    const float* norm_w     = (const float*)d_in[1];   // [2048]
    const float* in_proj_w  = (const float*)d_in[2];   // [6448,2048]
    const float* conv_w     = (const float*)d_in[3];   // [3072,4]
    const float* conv_b     = (const float*)d_in[4];   // [3072]
    const float* dt_bias    = (const float*)d_in[5];   // [48]
    const float* A_log      = (const float*)d_in[6];   // [48]
    const float* D_param    = (const float*)d_in[7];   // [48]
    const float* inner_w    = (const float*)d_in[8];   // [3072]
    const float* out_proj_w = (const float*)d_in[9];   // [2048,3072]
    const float* conv_state = (const float*)d_in[10];  // [1,3072,3]
    const float* ssm_state  = (const float*)d_in[11];  // [1,48,128,64]
    float* out = (float*)d_out;                        // [1,1024,2048]

    // workspace layout (all offsets multiple of 256 bytes)
    char* ws = (char*)d_ws;
    bf16_t* xn_bf = (bf16_t*)(ws);                       //  4,194,304  (1024x2048 bf16)
    bf16_t* w1_bf = (bf16_t*)(ws + 4194304);             // 26,476,544  (6464x2048 bf16, padded)
    float*  proj  = (float*) (ws + 30670848);            // 26,411,008  (1024x6448 f32)
    float*  xssm  = (float*) (ws + 57081856);            // 12,582,912  (1024x3072 f32)
    float*  yssm  = (float*) (ws + 69664768);            // 12,582,912  (1024x3072 f32)
    bf16_t* yg_bf = (bf16_t*)(ws + 82247680);            //  6,291,456  (1024x3072 bf16)
    bf16_t* w2_bf = (bf16_t*)(ws + 88539136);            // 12,582,912  (2048x3072 bf16)
                                                         // total ~101.1 MB

    // 1) input RMSNorm -> bf16
    rmsnorm_x_kernel<<<SEQ, 256, 0, stream>>>(x, norm_w, xn_bf);

    // 2) weight conversions (w1 padded to 6464 rows, pad rows zeroed)
    cvt_bf16_kernel<<<(D_PROJ_PAD * DIM + 255) / 256, 256, 0, stream>>>(
        in_proj_w, w1_bf, D_PROJ, DIM, D_PROJ_PAD);
    cvt_bf16_kernel<<<(DIM * D_INNER + 255) / 256, 256, 0, stream>>>(
        out_proj_w, w2_bf, DIM, D_INNER, DIM);

    // 3) in_proj GEMM: [1024,6448] = xn[1024,2048] * W1[6448,2048]^T
    gemm_bf16_nt_kernel<<<dim3(D_PROJ_PAD / 32, SEQ / 256), 128, 0, stream>>>(
        xn_bf, w1_bf, nullptr, proj, SEQ, D_PROJ, DIM);

    // 4) depthwise causal conv + SiLU
    conv_silu_kernel<<<dim3(D_INNER / 256, SEQ), 256, 0, stream>>>(
        proj, conv_w, conv_b, conv_state, xssm);

    // 5) SSM scan (96 workgroups: 48 heads x 2 p-halves)
    ssm_scan_kernel<<<N_HEADS * 2, 256, 0, stream>>>(
        proj, xssm, dt_bias, A_log, D_param, ssm_state, yssm);

    // 6) inner RMSNorm + SiLU gate -> bf16
    gated_norm_kernel<<<SEQ, 256, 0, stream>>>(yssm, proj, inner_w, yg_bf);

    // 7) out_proj GEMM + residual: out = x + yg[1024,3072] * W2[2048,3072]^T
    gemm_bf16_nt_kernel<<<dim3(DIM / 32, SEQ / 256), 128, 0, stream>>>(
        yg_bf, w2_bf, x, out, SEQ, DIM, D_INNER);
}
